// MoE_17136919511844
// MI455X (gfx1250) — compile-verified, matrix operand order checked by reading, not run on hardware
//
#include <hip/hip_runtime.h>

#define NUM_E   8
#define TOPK    2
#define HID     2048
#define FFN     5632
#define NTOK    2048                 // BATCH * SEQ
#define NASSIGN (NTOK * TOPK)

typedef __attribute__((ext_vector_type(16))) __bf16 v16bf;
typedef __attribute__((ext_vector_type(8)))  float  v8f;
typedef __attribute__((ext_vector_type(4)))  int    v4i;

union Frag { uint4 u[2]; v16bf v; };
union F4   { float4 v; float f[4]; };

#if defined(__gfx1250__) && __has_builtin(__builtin_amdgcn_global_load_async_to_lds_b128)
#define MOE_ASYNC 1
#else
#define MOE_ASYNC 0
#endif

__device__ __forceinline__ unsigned short f2bf(float f) {
    unsigned u = __float_as_uint(f);
    u += 0x7fffu + ((u >> 16) & 1u);          // round-to-nearest-even
    return (unsigned short)(u >> 16);
}

__device__ __forceinline__ unsigned pack2bf(float lo, float hi) {
    return (unsigned)f2bf(lo) | ((unsigned)f2bf(hi) << 16);
}

#if MOE_ASYNC
__device__ __forceinline__ void async16(const void* g, void* l) {
    __builtin_amdgcn_global_load_async_to_lds_b128(
        (__attribute__((address_space(1))) v4i*)g,
        (__attribute__((address_space(3))) v4i*)l, 0, 0);
}
#endif

__device__ __forceinline__ void wait_async0() {
#if MOE_ASYNC
#if __has_builtin(__builtin_amdgcn_s_wait_asynccnt)
    __builtin_amdgcn_s_wait_asynccnt(0);
#else
    asm volatile("s_wait_asynccnt 0" ::: "memory");
#endif
#endif
}

// ---------------------------------------------------------------------------
// Router: logits = x @ gate_w, softmax, top-2, renormalize.
// Fused: converts the token's x row to bf16 for the WMMA GEMMs.
// ---------------------------------------------------------------------------
__global__ __launch_bounds__(256) void moe_router_kernel(
    const float* __restrict__ x, const float* __restrict__ gw,
    unsigned short* __restrict__ xbf,
    int* __restrict__ tidx, float* __restrict__ tw)
{
    const int t   = blockIdx.x;
    const int tid = threadIdx.x;
    __shared__ float red[256][NUM_E];

    float acc[NUM_E];
#pragma unroll
    for (int e = 0; e < NUM_E; ++e) acc[e] = 0.0f;

    const float* xr = x + (size_t)t * HID;
    for (int h = tid; h < HID; h += 256) {
        float xv = xr[h];
        xbf[(size_t)t * HID + h] = f2bf(xv);
        const float* g = gw + (size_t)h * NUM_E;
#pragma unroll
        for (int e = 0; e < NUM_E; ++e) acc[e] += xv * g[e];
    }
#pragma unroll
    for (int e = 0; e < NUM_E; ++e) red[tid][e] = acc[e];
    __syncthreads();
    for (int s = 128; s > 0; s >>= 1) {
        if (tid < s) {
#pragma unroll
            for (int e = 0; e < NUM_E; ++e) red[tid][e] += red[tid + s][e];
        }
        __syncthreads();
    }
    if (tid == 0) {
        float l[NUM_E];
#pragma unroll
        for (int e = 0; e < NUM_E; ++e) l[e] = red[0][e];
        float m = l[0];
#pragma unroll
        for (int e = 1; e < NUM_E; ++e) m = fmaxf(m, l[e]);
        float p[NUM_E];
#pragma unroll
        for (int e = 0; e < NUM_E; ++e) p[e] = __expf(l[e] - m);
        int i0 = 0;
#pragma unroll
        for (int e = 1; e < NUM_E; ++e) if (p[e] > p[i0]) i0 = e;
        int i1 = (i0 == 0) ? 1 : 0;
#pragma unroll
        for (int e = 0; e < NUM_E; ++e) if (e != i0 && p[e] > p[i1]) i1 = e;
        float inv = __builtin_amdgcn_rcpf(p[i0] + p[i1]);
        tidx[t * 2 + 0] = i0;  tw[t * 2 + 0] = p[i0] * inv;
        tidx[t * 2 + 1] = i1;  tw[t * 2 + 1] = p[i1] * inv;
    }
}

// ---------------------------------------------------------------------------
// Expert-assignment bookkeeping
// ---------------------------------------------------------------------------
__global__ void moe_zero_kernel(int* counts, int* cursors) {
    int t = threadIdx.x;
    if (t < NUM_E) { counts[t] = 0; cursors[t] = 0; }
}

__global__ __launch_bounds__(256) void moe_count_kernel(
    const int* __restrict__ tidx, int* __restrict__ counts)
{
    int t = blockIdx.x * 256 + threadIdx.x;
    if (t < NTOK) {
        atomicAdd(&counts[tidx[t * 2 + 0]], 1);
        atomicAdd(&counts[tidx[t * 2 + 1]], 1);
    }
}

__global__ void moe_scan_kernel(const int* __restrict__ counts, int* __restrict__ offs) {
    if (threadIdx.x == 0) {
        int s = 0;
        for (int e = 0; e < NUM_E; ++e) { offs[e] = s; s += counts[e]; }
    }
}

__global__ __launch_bounds__(256) void moe_assign_kernel(
    const int* __restrict__ tidx, const float* __restrict__ tw,
    const int* __restrict__ offs, int* __restrict__ cursors,
    int* __restrict__ slot_token, int* __restrict__ slot_k, float* __restrict__ slot_w)
{
    int t = blockIdx.x * 256 + threadIdx.x;
    if (t >= NTOK) return;
#pragma unroll
    for (int k = 0; k < TOPK; ++k) {
        int e    = tidx[t * 2 + k];
        int pos  = atomicAdd(&cursors[e], 1);     // order irrelevant: rows independent
        int slot = offs[e] + pos;
        slot_token[slot] = t;
        slot_k[slot]     = k;
        slot_w[slot]     = tw[t * 2 + k];
    }
}

// ---------------------------------------------------------------------------
// GEMM1: per expert, h1 = silu(x@w1) * (x@w3).  M=cnt, N=FFN, K=HID.
// 128x64 block tile, 8 waves.  Double-buffered pipeline:
//   - A tile (128x32 bf16, gathered token rows) copied once per chunk via
//     ASYNCcnt-tracked global_load_async_to_lds_b128 (2 x 16B per thread).
//   - B chunks: float4 register staging -> packed bf16 ds_store_b32,
//     K-transposed (+8 pad) so lane fragments are two 16B LDS loads.
//   - s_wait_asynccnt 0 + barrier fences chunk kk before WMMAs consume it,
//     while chunk kk+32 asyncs/loads fly during compute.
// ---------------------------------------------------------------------------
__global__ __launch_bounds__(256) void moe_ffn1_kernel(
    const unsigned short* __restrict__ xbf,
    const float* __restrict__ w1, const float* __restrict__ w3,
    const int* __restrict__ counts, const int* __restrict__ offs,
    const int* __restrict__ slot_token, unsigned short* __restrict__ h1)
{
    const int e   = blockIdx.z;
    const int cnt = counts[e];
    const int m_base = blockIdx.x * 128;          // x fastest: m-tiles share B column -> L2 reuse
    if (m_base >= cnt) return;
    const int off    = offs[e];
    const int n_base = blockIdx.y * 64;

    const int tid  = threadIdx.x;
    const int lane = tid & 31;
    const int wid  = tid >> 5;
    const int wm   = wid >> 2;                    // 0..1 : 64-row half
    const int wn   = wid & 3;                     // 0..3 : 16-col strip
    const int lrow = lane & 15;
    const int lhi  = lane >> 4;
    const int klo  = lhi << 3;

    __shared__ __align__(16) unsigned short lb1[2][64][40];
    __shared__ __align__(16) unsigned short lb3[2][64][40];

#if MOE_ASYNC
    __shared__ __align__(16) unsigned short la[2][128][40];
    const int arow_t = tid >> 1;                  // row 0..127 of the tile
    const int aoff   = (tid & 1) << 4;            // element 0 / 16 within 32-elem row
    {
    }
    int rA    = m_base + arow_t;
    int slotA = off + (rA < cnt ? rA : 0);
    const unsigned short* asrc = xbf + (size_t)slot_token[slotA] * HID;
#else
    const unsigned short* arow[4];
#pragma unroll
    for (int i = 0; i < 4; ++i) {
        int r    = m_base + wm * 64 + i * 16 + lrow;
        int slot = off + (r < cnt ? r : 0);
        arow[i]  = xbf + (size_t)slot_token[slot] * HID;
    }
#endif

    const float* B1 = w1 + (size_t)e * HID * FFN;
    const float* B3 = w3 + (size_t)e * HID * FFN;

    // B staging map: thread -> rows (2*kp, 2*kp+1), cols cp..cp+3 of the 32x64 chunk
    const int kp = tid >> 4;
    const int cp = (tid & 15) << 2;

    F4 r1a, r1b, r3a, r3b;
    auto loadB = [&](int kk) {
        const float* q1 = B1 + (size_t)(kk + 2 * kp) * FFN + (n_base + cp);
        const float* q3 = B3 + (size_t)(kk + 2 * kp) * FFN + (n_base + cp);
        r1a.v = *(const float4*)q1;
        r1b.v = *(const float4*)(q1 + FFN);
        r3a.v = *(const float4*)q3;
        r3b.v = *(const float4*)(q3 + FFN);
    };

    v8f acc1[4] = {};
    v8f acc3[4] = {};

#if MOE_ASYNC
    async16(asrc + aoff,     &la[0][arow_t][aoff]);
    async16(asrc + aoff + 8, &la[0][arow_t][aoff + 8]);
#endif
    loadB(0);
    int cur = 0;
    for (int kk = 0; kk < HID; kk += 32) {
        // commit staged B registers (chunk kk) into LDS buffer `cur`
#pragma unroll
        for (int c = 0; c < 4; ++c) {
            *(unsigned*)&lb1[cur][cp + c][2 * kp] = pack2bf(r1a.f[c], r1b.f[c]);
            *(unsigned*)&lb3[cur][cp + c][2 * kp] = pack2bf(r3a.f[c], r3b.f[c]);
        }
        wait_async0();                             // own chunk-kk A transfers landed
        __syncthreads();                           // everyone's landed; lb[cur] visible

        if (kk + 32 < HID) {
#if MOE_ASYNC
            const unsigned short* an = asrc + (kk + 32) + aoff;
            async16(an,     &la[cur ^ 1][arow_t][aoff]);
            async16(an + 8, &la[cur ^ 1][arow_t][aoff + 8]);
#endif
            loadB(kk + 32);                        // in flight during WMMAs
            if (kk + 64 < HID) {
                __builtin_prefetch(B1 + (size_t)(kk + 64 + 2 * kp) * FFN + (n_base + cp), 0, 1);
                __builtin_prefetch(B3 + (size_t)(kk + 64 + 2 * kp) * FFN + (n_base + cp), 0, 1);
            }
        }

        const unsigned short* bc1 = &lb1[cur][wn * 16 + lrow][0];
        const unsigned short* bc3 = &lb3[cur][wn * 16 + lrow][0];
        Frag bf1, bf3;
        bf1.u[0] = *(const uint4*)(bc1 + klo);
        bf1.u[1] = *(const uint4*)(bc1 + klo + 16);
        bf3.u[0] = *(const uint4*)(bc3 + klo);
        bf3.u[1] = *(const uint4*)(bc3 + klo + 16);

#pragma unroll
        for (int i = 0; i < 4; ++i) {
            Frag a;
#if MOE_ASYNC
            const unsigned short* ap = &la[cur][wm * 64 + i * 16 + lrow][0];
            a.u[0] = *(const uint4*)(ap + klo);
            a.u[1] = *(const uint4*)(ap + klo + 16);
#else
            a.u[0] = *(const uint4*)(arow[i] + kk + klo);
            a.u[1] = *(const uint4*)(arow[i] + kk + klo + 16);
#endif
            acc1[i] = __builtin_amdgcn_wmma_f32_16x16x32_bf16(
                false, a.v, false, bf1.v, (short)0, acc1[i], false, false);
            acc3[i] = __builtin_amdgcn_wmma_f32_16x16x32_bf16(
                false, a.v, false, bf3.v, (short)0, acc3[i], false, false);
        }
        cur ^= 1;
    }

    const int colg = n_base + wn * 16 + lrow;
#pragma unroll
    for (int i = 0; i < 4; ++i) {
        int rsub = m_base + wm * 64 + i * 16 + lhi * 8;
#pragma unroll
        for (int v = 0; v < 8; ++v) {
            int rg = rsub + v;
            if (rg < cnt) {
                float g1  = acc1[i][v];
                float g3  = acc3[i][v];
                float sig = __builtin_amdgcn_rcpf(1.0f + __expf(-g1));
                h1[(size_t)(off + rg) * FFN + colg] = f2bf(g1 * sig * g3);
            }
        }
    }
}

// ---------------------------------------------------------------------------
// GEMM2: per expert, out_pa[(k, token)] = weight * (h1 @ w2).  M=cnt, N=HID, K=FFN.
// Same pipeline; A rows (h1) are already bf16 -> pure async copy, no convert.
// ---------------------------------------------------------------------------
__global__ __launch_bounds__(256) void moe_ffn2_kernel(
    const unsigned short* __restrict__ h1, const float* __restrict__ w2,
    const int* __restrict__ counts, const int* __restrict__ offs,
    const int* __restrict__ slot_token, const int* __restrict__ slot_k,
    const float* __restrict__ slot_w, float* __restrict__ out_pa)
{
    const int e   = blockIdx.z;
    const int cnt = counts[e];
    const int m_base = blockIdx.x * 128;
    if (m_base >= cnt) return;
    const int off    = offs[e];
    const int n_base = blockIdx.y * 64;

    const int tid  = threadIdx.x;
    const int lane = tid & 31;
    const int wid  = tid >> 5;
    const int wm   = wid >> 2;
    const int wn   = wid & 3;
    const int lrow = lane & 15;
    const int lhi  = lane >> 4;
    const int klo  = lhi << 3;

    __shared__ __align__(16) unsigned short lb2[2][64][40];

#if MOE_ASYNC
    __shared__ __align__(16) unsigned short la[2][128][40];
    const int arow_t = tid >> 1;
    const int aoff   = (tid & 1) << 4;
    int rA    = m_base + arow_t;
    int slotA = off + (rA < cnt ? rA : 0);
    const unsigned short* asrc = h1 + (size_t)slotA * FFN;
#else
    const unsigned short* arow[4];
#pragma unroll
    for (int i = 0; i < 4; ++i) {
        int r    = m_base + wm * 64 + i * 16 + lrow;
        int slot = off + (r < cnt ? r : 0);
        arow[i]  = h1 + (size_t)slot * FFN;
    }
#endif

    const float* B2 = w2 + (size_t)e * FFN * HID;

    const int kp = tid >> 4;
    const int cp = (tid & 15) << 2;

    F4 r2a, r2b;
    auto loadB = [&](int kk) {
        const float* q2 = B2 + (size_t)(kk + 2 * kp) * HID + (n_base + cp);
        r2a.v = *(const float4*)q2;
        r2b.v = *(const float4*)(q2 + HID);
    };

    v8f acc[4] = {};

#if MOE_ASYNC
    async16(asrc + aoff,     &la[0][arow_t][aoff]);
    async16(asrc + aoff + 8, &la[0][arow_t][aoff + 8]);
#endif
    loadB(0);
    int cur = 0;
    for (int kk = 0; kk < FFN; kk += 32) {
#pragma unroll
        for (int c = 0; c < 4; ++c) {
            *(unsigned*)&lb2[cur][cp + c][2 * kp] = pack2bf(r2a.f[c], r2b.f[c]);
        }
        wait_async0();
        __syncthreads();

        if (kk + 32 < FFN) {
#if MOE_ASYNC
            const unsigned short* an = asrc + (kk + 32) + aoff;
            async16(an,     &la[cur ^ 1][arow_t][aoff]);
            async16(an + 8, &la[cur ^ 1][arow_t][aoff + 8]);
#endif
            loadB(kk + 32);
            if (kk + 64 < FFN) {
                __builtin_prefetch(B2 + (size_t)(kk + 64 + 2 * kp) * HID + (n_base + cp), 0, 1);
            }
        }

        const unsigned short* bc = &lb2[cur][wn * 16 + lrow][0];
        Frag b;
        b.u[0] = *(const uint4*)(bc + klo);
        b.u[1] = *(const uint4*)(bc + klo + 16);

#pragma unroll
        for (int i = 0; i < 4; ++i) {
            Frag a;
#if MOE_ASYNC
            const unsigned short* ap = &la[cur][wm * 64 + i * 16 + lrow][0];
            a.u[0] = *(const uint4*)(ap + klo);
            a.u[1] = *(const uint4*)(ap + klo + 16);
#else
            a.u[0] = *(const uint4*)(arow[i] + kk + klo);
            a.u[1] = *(const uint4*)(arow[i] + kk + klo + 16);
#endif
            acc[i] = __builtin_amdgcn_wmma_f32_16x16x32_bf16(
                false, a.v, false, b.v, (short)0, acc[i], false, false);
        }
        cur ^= 1;
    }

    const int colg = n_base + wn * 16 + lrow;
#pragma unroll
    for (int i = 0; i < 4; ++i) {
        int rsub = m_base + wm * 64 + i * 16 + lhi * 8;
#pragma unroll
        for (int v = 0; v < 8; ++v) {
            int rg = rsub + v;
            if (rg < cnt) {
                int   slot = off + rg;
                int   t    = slot_token[slot];
                int   kx   = slot_k[slot];
                float wgt  = slot_w[slot];
                out_pa[((size_t)kx * NTOK + t) * HID + colg] = wgt * acc[i][v];
            }
        }
    }
}

// ---------------------------------------------------------------------------
// Combine: out[t] = pa[k=0][t] + pa[k=1][t]   (deterministic, no atomics)
// ---------------------------------------------------------------------------
__global__ __launch_bounds__(256) void moe_combine_kernel(
    const float* __restrict__ pa, float* __restrict__ out)
{
    size_t i = (size_t)blockIdx.x * 256 + threadIdx.x;   // over NTOK*HID/4
    const float4* a = (const float4*)pa;
    const float4* b = (const float4*)(pa + (size_t)NTOK * HID);
    float4 va = a[i], vb = b[i];
    float4 r;
    r.x = va.x + vb.x; r.y = va.y + vb.y; r.z = va.z + vb.z; r.w = va.w + vb.w;
    ((float4*)out)[i] = r;
}

// ---------------------------------------------------------------------------
extern "C" void kernel_launch(void* const* d_in, const int* in_sizes, int n_in,
                              void* d_out, int out_size, void* d_ws, size_t ws_size,
                              hipStream_t stream) {
    const float* x  = (const float*)d_in[0];
    const float* gw = (const float*)d_in[1];
    const float* w1 = (const float*)d_in[2];
    const float* w3 = (const float*)d_in[3];
    const float* w2 = (const float*)d_in[4];
    float* out = (float*)d_out;

    char* p = (char*)d_ws;
    auto carve = [&](size_t bytes) -> void* {
        void* r = (void*)p;
        p += (bytes + 255) & ~(size_t)255;
        return r;
    };

    unsigned short* xbf     = (unsigned short*)carve((size_t)NTOK * HID * 2);
    int*            tidx    = (int*)  carve((size_t)NTOK * TOPK * 4);
    float*          tw      = (float*)carve((size_t)NTOK * TOPK * 4);
    int*            counts  = (int*)  carve(NUM_E * 4);
    int*            cursors = (int*)  carve(NUM_E * 4);
    int*            offs    = (int*)  carve(NUM_E * 4);
    int*            s_tok   = (int*)  carve((size_t)NASSIGN * 4);
    int*            s_k     = (int*)  carve((size_t)NASSIGN * 4);
    float*          s_w     = (float*)carve((size_t)NASSIGN * 4);
    unsigned short* h1      = (unsigned short*)carve((size_t)NASSIGN * FFN * 2);
    float*          out_pa  = (float*)carve((size_t)2 * NTOK * HID * 4);

    moe_router_kernel<<<NTOK, 256, 0, stream>>>(x, gw, xbf, tidx, tw);
    moe_zero_kernel<<<1, 32, 0, stream>>>(counts, cursors);
    moe_count_kernel<<<NTOK / 256, 256, 0, stream>>>(tidx, counts);
    moe_scan_kernel<<<1, 32, 0, stream>>>(counts, offs);
    moe_assign_kernel<<<NTOK / 256, 256, 0, stream>>>(tidx, tw, offs, cursors, s_tok, s_k, s_w);

    dim3 g1(NTOK / 128, FFN / 64, NUM_E);
    moe_ffn1_kernel<<<g1, 256, 0, stream>>>(xbf, w1, w3, counts, offs, s_tok, h1);

    dim3 g2(NTOK / 128, HID / 64, NUM_E);
    moe_ffn2_kernel<<<g2, 256, 0, stream>>>(h1, w2, counts, offs, s_tok, s_k, s_w, out_pa);

    moe_combine_kernel<<<(NTOK * HID / 4) / 256, 256, 0, stream>>>(out_pa, out);
}
